// RMVSNet_15693810499598
// MI455X (gfx1250) — matrix operand order repaired and checked by reading, not compiled, not run on hardware
//
#include <hip/hip_runtime.h>
#include <hip/hip_bf16.h>

typedef __attribute__((ext_vector_type(16))) _Float16 v16h;
typedef __attribute__((ext_vector_type(8)))  float    v8f;

#define ACT_NONE 0
#define ACT_RELU 1
#define ACT_SIG  2
#define ACT_TANH 3

#define MAX_KP 448   // max padded K: 48ch * 9 = 432 -> 448

// ---------------------------------------------------------------------------
// Implicit-GEMM conv (NCHW, OIHW weights, explicit pad, stride s).
// Block = 128 threads (4 waves). Each wave: 32 output pixels x 16 couts
// (two v8f accumulators sharing one B fragment). Weights for the block's
// 16-cout tile are staged once into LDS as f16; B fragments are contiguous
// 32-byte LDS reads. K/stride are template constants so im2col address
// decode is strength-reduced (no runtime idiv).
//   A frag (16x32 f16): lane = M, VGPR i packs K=16*half+2i+j  (contiguous 16)
//   B frag (32x16 f16): lane = N, VGPR i packs K=16*half+2i+j  (contiguous 16)
//   C/D: VGPR r -> M=r+8*half, N=lane%16
// ---------------------------------------------------------------------------
template <int KSZ, int STRIDE>
__global__ __launch_bounds__(128) void conv_wmma(
    const float* __restrict__ in,    // [N, Cin, H, W]
    const float* __restrict__ wgt,   // [Cout, Cin, KSZ, KSZ]
    const float* __restrict__ bias,  // [Cout]
    float* __restrict__ out,         // [N, Cout, Ho, Wo]
    int Cin, int Cout, int KP, int padH, int padW,
    int H, int W, int Ho, int Wo, int act)
{
    constexpr int KK = KSZ * KSZ;
    const int Ktot = Cin * KK;

    __shared__ __attribute__((aligned(32))) _Float16 wtile[16 * MAX_KP];

    const int tid  = threadIdx.x;
    const int lane = tid & 31;
    const int wave = tid >> 5;
    const int half = lane >> 4;
    const int lr   = lane & 15;
    const int nTile = blockIdx.y;
    const int b     = blockIdx.z;
    const int M     = Ho * Wo;

    // ---- stage 16-cout weight tile into LDS (f32 -> f16), zero-padded ----
    for (int n = 0; n < 16; ++n) {
        int co = nTile * 16 + n;
        for (int k = tid; k < KP; k += 128) {
            float v = (co < Cout && k < Ktot) ? wgt[(size_t)co * Ktot + k] : 0.f;
            wtile[n * KP + k] = (_Float16)v;
        }
    }
    __syncthreads();

    const size_t HWin = (size_t)H * W;
    const float* inB = in + (size_t)b * Cin * HWin;

    const int mBase = (blockIdx.x * 4 + wave) * 32;
    const int m0 = mBase + lr;
    const int m1 = mBase + 16 + lr;
    const bool mOK0 = (m0 < M);
    const bool mOK1 = (m1 < M);
    const int oy0 = mOK0 ? (m0 / Wo) : 0;
    const int ox0 = mOK0 ? (m0 % Wo) : 0;
    const int oy1 = mOK1 ? (m1 / Wo) : 0;
    const int ox1 = mOK1 ? (m1 % Wo) : 0;

    const int  n   = nTile * 16 + lr;
    const bool nOK = (n < Cout);

    v8f acc0 = {}, acc1 = {};
    for (int k0 = 0; k0 < Ktot; k0 += 32) {
        v16h a0, a1;
#pragma unroll
        for (int i = 0; i < 8; ++i) {
#pragma unroll
            for (int j = 0; j < 2; ++j) {
                int ka = k0 + half * 16 + (i << 1) + j;
                float av0 = 0.f, av1 = 0.f;
                if (ka < Ktot) {
                    int ci = ka / KK;           // constant divisor -> mul/shift
                    int rr = ka - ci * KK;
                    int ky = rr / KSZ;
                    int kx = rr - ky * KSZ;
                    if (mOK0) {
                        int iy = oy0 * STRIDE + ky - padH;
                        int ix = ox0 * STRIDE + kx - padW;
                        if ((unsigned)iy < (unsigned)H && (unsigned)ix < (unsigned)W)
                            av0 = inB[ci * HWin + (size_t)iy * W + ix];
                    }
                    if (mOK1) {
                        int iy = oy1 * STRIDE + ky - padH;
                        int ix = ox1 * STRIDE + kx - padW;
                        if ((unsigned)iy < (unsigned)H && (unsigned)ix < (unsigned)W)
                            av1 = inB[ci * HWin + (size_t)iy * W + ix];
                    }
                }
                a0[i * 2 + j] = (_Float16)av0;
                a1[i * 2 + j] = (_Float16)av1;
            }
        }
        // B fragment: 16 contiguous f16 from LDS (32B aligned by construction)
        v16h bf = *(const v16h*)&wtile[lr * KP + k0 + half * 16];
        acc0 = __builtin_amdgcn_wmma_f32_16x16x32_f16(
            false, a0, false, bf, (short)0, acc0, false, false);
        acc1 = __builtin_amdgcn_wmma_f32_16x16x32_f16(
            false, a1, false, bf, (short)0, acc1, false, false);
    }

    if (nOK) {
        float bs = bias[n];
#pragma unroll
        for (int r = 0; r < 8; ++r) {
            int mr0 = mBase + r + 8 * half;
            int mr1 = mr0 + 16;
            if (mr0 < M) {
                float v = acc0[r] + bs;
                if      (act == ACT_RELU) v = fmaxf(v, 0.f);
                else if (act == ACT_SIG)  v = 1.f / (1.f + __expf(-v));
                else if (act == ACT_TANH) v = tanhf(v);
                out[((size_t)b * Cout + n) * M + mr0] = v;
            }
            if (mr1 < M) {
                float v = acc1[r] + bs;
                if      (act == ACT_RELU) v = fmaxf(v, 0.f);
                else if (act == ACT_SIG)  v = 1.f / (1.f + __expf(-v));
                else if (act == ACT_TANH) v = tanhf(v);
                out[((size_t)b * Cout + n) * M + mr1] = v;
            }
        }
    }
}

// ---------------------------------------------------------------------------
// P_rel[v] = projections[v+1] @ inv(projections[0])  (single-thread, 4x4)
// ---------------------------------------------------------------------------
__global__ void prel_kernel(const float* __restrict__ proj, float* __restrict__ Prel)
{
    if (threadIdx.x != 0 || blockIdx.x != 0) return;
    float a[4][8];
    for (int i = 0; i < 4; ++i)
        for (int j = 0; j < 4; ++j) {
            a[i][j]     = proj[i * 4 + j];
            a[i][4 + j] = (i == j) ? 1.f : 0.f;
        }
    for (int c = 0; c < 4; ++c) {
        int p = c;
        for (int r = c + 1; r < 4; ++r)
            if (fabsf(a[r][c]) > fabsf(a[p][c])) p = r;
        if (p != c)
            for (int j = 0; j < 8; ++j) { float t = a[c][j]; a[c][j] = a[p][j]; a[p][j] = t; }
        float inv = 1.f / a[c][c];
        for (int j = 0; j < 8; ++j) a[c][j] *= inv;
        for (int r = 0; r < 4; ++r) {
            if (r == c) continue;
            float f = a[r][c];
            for (int j = 0; j < 8; ++j) a[r][j] -= f * a[c][j];
        }
    }
    for (int v = 0; v < 2; ++v)
        for (int i = 0; i < 4; ++i)
            for (int j = 0; j < 4; ++j) {
                float s = 0.f;
                for (int k = 0; k < 4; ++k)
                    s += proj[(1 + v) * 16 + i * 4 + k] * a[k][4 + j];
                Prel[v * 16 + i * 4 + j] = s;
            }
}

// ---------------------------------------------------------------------------
// Homography warp (bilinear, zero outside) + variance cost across 3 views.
// Writes -cost (GRU1 input), [Cf][h][w].
// ---------------------------------------------------------------------------
__global__ void cost_kernel(const float* __restrict__ fref,   // [Cf][h][w]
                            const float* __restrict__ fsrc,   // [2][Cf][h][w]
                            const float* __restrict__ Prel,   // [2][4][4]
                            const float* __restrict__ depths, int dIdx,
                            float* __restrict__ out, int Cf, int h, int w)
{
    int idx = blockIdx.x * blockDim.x + threadIdx.x;
    int HW = h * w;
    if (idx >= HW) return;
    float d  = depths[dIdx];
    float px = (float)(idx % w);
    float py = (float)(idx / w);

    int   offs[2][4];
    float wts[2][4];
    for (int v = 0; v < 2; ++v) {
        const float* P = Prel + v * 16;
        float X0 = px * d, X1 = py * d, X2 = d;
        float w0 = P[0] * X0 + P[1] * X1 + P[2]  * X2 + P[3];
        float w1 = P[4] * X0 + P[5] * X1 + P[6]  * X2 + P[7];
        float w2 = P[8] * X0 + P[9] * X1 + P[10] * X2 + P[11];
        float x = w0 / (w2 + 1e-6f);
        float y = w1 / (w2 + 1e-6f);
        float x0f = floorf(x), y0f = floorf(y);
        float wx = x - x0f, wy = y - y0f;
        float tw[4] = { (1.f - wx) * (1.f - wy), wx * (1.f - wy),
                        (1.f - wx) * wy,         wx * wy };
#pragma unroll
        for (int t = 0; t < 4; ++t) {
            float xi = x0f + (float)(t & 1);
            float yi = y0f + (float)(t >> 1);
            bool valid = (xi >= 0.f) && (xi <= (float)(w - 1)) &&
                         (yi >= 0.f) && (yi <= (float)(h - 1));
            int xc = min(max((int)xi, 0), w - 1);
            int yc = min(max((int)yi, 0), h - 1);
            offs[v][t] = yc * w + xc;
            wts[v][t]  = valid ? tw[t] : 0.f;
        }
    }
    for (int c = 0; c < Cf; ++c) {
        float f0    = fref[(size_t)c * HW + idx];
        float sum   = f0;
        float sumsq = f0 * f0;
        for (int v = 0; v < 2; ++v) {
            const float* img = fsrc + ((size_t)v * Cf + c) * HW;
            float fv = wts[v][0] * img[offs[v][0]] + wts[v][1] * img[offs[v][1]]
                     + wts[v][2] * img[offs[v][2]] + wts[v][3] * img[offs[v][3]];
            sum   += fv;
            sumsq += fv * fv;
        }
        float mean = sum * (1.f / 3.f);
        out[(size_t)c * HW + idx] = -(sumsq * (1.f / 3.f) - mean * mean);
    }
}

__global__ void zero_kernel(float* __restrict__ p, int n)
{
    int i = blockIdx.x * blockDim.x + threadIdx.x;
    if (i < n) p[i] = 0.f;
}

__global__ void ewise_mul(const float* __restrict__ a, const float* __restrict__ b,
                          float* __restrict__ dst, int n)
{
    int i = blockIdx.x * blockDim.x + threadIdx.x;
    if (i < n) dst[i] = a[i] * b[i];
}

__global__ void gru_update(float* __restrict__ hbuf, const float* __restrict__ z,
                           const float* __restrict__ c, int n)
{
    int i = blockIdx.x * blockDim.x + threadIdx.x;
    if (i < n) hbuf[i] = (1.f - z[i]) * hbuf[i] + z[i] * c[i];
}

__global__ void softmax_depth(const float* __restrict__ costs, // [D][HW]
                              const float* __restrict__ depths,
                              float* __restrict__ out, int D, int HW)
{
    int idx = blockIdx.x * blockDim.x + threadIdx.x;
    if (idx >= HW) return;
    float mx = -1e30f;
    for (int d = 0; d < D; ++d) mx = fmaxf(mx, costs[(size_t)d * HW + idx]);
    float s = 0.f, acc = 0.f;
    for (int d = 0; d < D; ++d) {
        float e = __expf(costs[(size_t)d * HW + idx] - mx);
        s += e;
        acc += e * depths[d];
    }
    out[idx] = acc / s;
}

// ---------------------------------------------------------------------------
extern "C" void kernel_launch(void* const* d_in, const int* in_sizes, int n_in,
                              void* d_out, int out_size, void* d_ws, size_t ws_size,
                              hipStream_t stream)
{
    (void)in_sizes; (void)n_in; (void)out_size; (void)ws_size;

    // ---- inputs (setup_inputs insertion order; params flattened in order) ----
    const float* images = (const float*)d_in[0];   // (3,3,512,640)
    const float* proj   = (const float*)d_in[1];   // (1,3,4,4)
    const float* depths = (const float*)d_in[2];   // (1,64)
    const float* fe_w[8], *fe_b[8];
    for (int i = 0; i < 8; ++i) fe_w[i] = (const float*)d_in[3 + i];
    for (int i = 0; i < 8; ++i) fe_b[i] = (const float*)d_in[11 + i];
    const float* g1_wg = (const float*)d_in[19];
    const float* g1_bg = (const float*)d_in[20];
    const float* g1_wc = (const float*)d_in[21];
    const float* g1_bc = (const float*)d_in[22];
    const float* g2_wg = (const float*)d_in[23];
    const float* g2_bg = (const float*)d_in[24];
    const float* g2_wc = (const float*)d_in[25];
    const float* g2_bc = (const float*)d_in[26];
    const float* g3_wg = (const float*)d_in[27];
    const float* g3_bg = (const float*)d_in[28];
    const float* g3_wc = (const float*)d_in[29];
    const float* g3_bc = (const float*)d_in[30];
    const float* cv_w  = (const float*)d_in[31];
    const float* cv_b  = (const float*)d_in[32];

    const int hF = 128, wF = 160, HW = hF * wF, D = 64, Cf = 32;

    // ---- workspace carving ----
    float* ws = (float*)d_ws;
    size_t off = 0;
    auto carve = [&](size_t n) { float* p = ws + off; off += n; return p; };
    float* bufA  = carve((size_t)3 * 8 * 512 * 640);   // FE ping
    float* bufB  = carve((size_t)3 * 8 * 512 * 640);   // FE pong
    float* feats = carve((size_t)3 * Cf * HW);         // persistent features
    float* Prel  = carve(32);
    float* ncost = carve((size_t)32 * HW);
    float* h1    = carve((size_t)16 * HW);
    float* h2    = carve((size_t)4  * HW);
    float* h3    = carve((size_t)2  * HW);
    float* cat   = carve((size_t)48 * HW);
    float* gbuf  = carve((size_t)32 * HW);
    float* cbuf  = carve((size_t)16 * HW);
    float* costs = carve((size_t)D * HW);

    auto launch_conv = [&](const float* in, const float* w, const float* b, float* out,
                           int N, int Cin, int Cout, int K, int s, int Hin, int Win, int act) {
        int Ho = (Hin + s - 1) / s;
        int Wo = (Win + s - 1) / s;
        int padH = ((Ho - 1) * s + K - Hin) / 2; if (padH < 0) padH = 0;
        int padW = ((Wo - 1) * s + K - Win) / 2; if (padW < 0) padW = 0;
        int M  = Ho * Wo;
        int Ktot = Cin * K * K;
        int KP = ((Ktot + 31) / 32) * 32;
        dim3 grid((M + 127) / 128, (Cout + 15) / 16, N);
        if (K == 5 && s == 2)
            conv_wmma<5, 2><<<grid, 128, 0, stream>>>(in, w, b, out, Cin, Cout, KP,
                                                      padH, padW, Hin, Win, Ho, Wo, act);
        else
            conv_wmma<3, 1><<<grid, 128, 0, stream>>>(in, w, b, out, Cin, Cout, KP,
                                                      padH, padW, Hin, Win, Ho, Wo, act);
    };

    // ---- feature extraction (3 images batched in grid.z) ----
    struct { int cin, cout, k, s; } cfg[8] = {
        {3, 8, 3, 1}, {8, 8, 3, 1}, {8, 16, 5, 2}, {16, 16, 3, 1},
        {16, 16, 3, 1}, {16, 32, 5, 2}, {32, 32, 3, 1}, {32, 32, 3, 1}};
    const float* cur = images;
    int Hc = 512, Wc = 640;
    for (int i = 0; i < 8; ++i) {
        float* outp = (i == 7) ? feats : ((i & 1) ? bufB : bufA);
        int act = (i == 7) ? ACT_NONE : ACT_RELU;
        launch_conv(cur, fe_w[i], fe_b[i], outp, 3, cfg[i].cin, cfg[i].cout,
                    cfg[i].k, cfg[i].s, Hc, Wc, act);
        Hc = (Hc + cfg[i].s - 1) / cfg[i].s;
        Wc = (Wc + cfg[i].s - 1) / cfg[i].s;
        cur = outp;
    }

    // ---- relative projections + zeroed GRU states ----
    prel_kernel<<<1, 1, 0, stream>>>(proj, Prel);
    {
        int n = (16 + 4 + 2) * HW;   // h1,h2,h3 are contiguous
        zero_kernel<<<(n + 255) / 256, 256, 0, stream>>>(h1, n);
    }

    const float* fref = feats;
    const float* fsrc = feats + (size_t)Cf * HW;
    const int EB = 256;

    // ---- recurrent depth sweep ----
    for (int d = 0; d < D; ++d) {
        cost_kernel<<<(HW + EB - 1) / EB, EB, 0, stream>>>(
            fref, fsrc, Prel, depths, d, ncost, Cf, hF, wF);

        // GRU1: x = -cost (32ch), h = h1 (16ch)
        hipMemcpyAsync(cat, ncost, (size_t)32 * HW * 4, hipMemcpyDeviceToDevice, stream);
        hipMemcpyAsync(cat + (size_t)32 * HW, h1, (size_t)16 * HW * 4,
                       hipMemcpyDeviceToDevice, stream);
        launch_conv(cat, g1_wg, g1_bg, gbuf, 1, 48, 32, 3, 1, hF, wF, ACT_SIG);
        ewise_mul<<<(16 * HW + EB - 1) / EB, EB, 0, stream>>>(
            gbuf, h1, cat + (size_t)32 * HW, 16 * HW);           // r*h
        launch_conv(cat, g1_wc, g1_bc, cbuf, 1, 48, 16, 3, 1, hF, wF, ACT_TANH);
        gru_update<<<(16 * HW + EB - 1) / EB, EB, 0, stream>>>(
            h1, gbuf + (size_t)16 * HW, cbuf, 16 * HW);

        // GRU2: x = h1 (16ch), h = h2 (4ch)
        hipMemcpyAsync(cat, h1, (size_t)16 * HW * 4, hipMemcpyDeviceToDevice, stream);
        hipMemcpyAsync(cat + (size_t)16 * HW, h2, (size_t)4 * HW * 4,
                       hipMemcpyDeviceToDevice, stream);
        launch_conv(cat, g2_wg, g2_bg, gbuf, 1, 20, 8, 3, 1, hF, wF, ACT_SIG);
        ewise_mul<<<(4 * HW + EB - 1) / EB, EB, 0, stream>>>(
            gbuf, h2, cat + (size_t)16 * HW, 4 * HW);
        launch_conv(cat, g2_wc, g2_bc, cbuf, 1, 20, 4, 3, 1, hF, wF, ACT_TANH);
        gru_update<<<(4 * HW + EB - 1) / EB, EB, 0, stream>>>(
            h2, gbuf + (size_t)4 * HW, cbuf, 4 * HW);

        // GRU3: x = h2 (4ch), h = h3 (2ch)
        hipMemcpyAsync(cat, h2, (size_t)4 * HW * 4, hipMemcpyDeviceToDevice, stream);
        hipMemcpyAsync(cat + (size_t)4 * HW, h3, (size_t)2 * HW * 4,
                       hipMemcpyDeviceToDevice, stream);
        launch_conv(cat, g3_wg, g3_bg, gbuf, 1, 6, 4, 3, 1, hF, wF, ACT_SIG);
        ewise_mul<<<(2 * HW + EB - 1) / EB, EB, 0, stream>>>(
            gbuf, h3, cat + (size_t)4 * HW, 2 * HW);
        launch_conv(cat, g3_wc, g3_bc, cbuf, 1, 6, 2, 3, 1, hF, wF, ACT_TANH);
        gru_update<<<(2 * HW + EB - 1) / EB, EB, 0, stream>>>(
            h3, gbuf + (size_t)2 * HW, cbuf, 2 * HW);

        // regularizer: 2 -> 1 channel, 3x3
        launch_conv(h3, cv_w, cv_b, costs + (size_t)d * HW, 1, 2, 1, 3, 1, hF, wF, ACT_NONE);
    }

    // ---- softmax over depth + expectation ----
    softmax_depth<<<(HW + EB - 1) / EB, EB, 0, stream>>>(
        costs, depths, (float*)d_out, D, HW);
}
